// MyModel_87522843558936
// MI455X (gfx1250) — matrix operand order repaired and checked by reading, not compiled, not run on hardware
//
#include <hip/hip_runtime.h>

typedef __attribute__((ext_vector_type(16))) _Float16 v16h;
typedef __attribute__((ext_vector_type(8)))  float    v8f;

#define VOCAB  100
#define EMBED  16
#define STATE  7
#define NOUT   10
#define BATCH  4096
#define TSTEPS 512
#define FOURS  (4 * STATE)   // 28 gate columns: [i 0-6 | f 7-13 | g 14-20 | o 21-27]

union B16u { v16h h; unsigned int u[8]; };

__device__ __forceinline__ float rcp_f(float x) { return __builtin_amdgcn_rcpf(x); }
__device__ __forceinline__ float sigmoid_f(float x) { return rcp_f(1.0f + __expf(-x)); }
__device__ __forceinline__ float tanh_f(float x) { return 1.0f - 2.0f * rcp_f(1.0f + __expf(2.0f * x)); }

__device__ __forceinline__ unsigned int pack2h(float a, float b) {
    union { _Float16 h[2]; unsigned int u; } p;
    p.h[0] = (_Float16)a;
    p.h[1] = (_Float16)b;
    return p.u;
}

// One wave32 per 16-row batch tile; transposed GEMM keeps the recurrence lane-local.
__global__ __launch_bounds__(32) void lstm_wmma_kernel(
    const int*   __restrict__ tokens,  // [B, T]
    const float* __restrict__ emb,     // [VOCAB, EMBED]
    const float* __restrict__ W,       // [EMBED, 4S]
    const float* __restrict__ U,       // [STATE, 4S]
    const float* __restrict__ b,       // [4S]
    const float* __restrict__ Wd,      // [STATE, NOUT]
    const float* __restrict__ bd,      // [NOUT]
    float*       __restrict__ out)     // [B, T, NOUT]
{
    __shared__ __align__(16) _Float16 s_emb[VOCAB * EMBED]; // f16 table, 3.2 KB

    const int lane = threadIdx.x;  // 0..31
    const int m    = lane & 15;    // batch row (B/D cols) | weight row j (A)
    const int half = lane >> 4;
    const int b0   = blockIdx.x * 16;

    // ---- one-time: embedding table -> LDS as f16 ----
    for (int i = lane; i < VOCAB * EMBED; i += 32)
        s_emb[i] = (_Float16)emb[i];
    __syncthreads();

    // ---- constant A operands: A_g[j,k] = [W_g^T | U_g^T], 16x32 f16 ----
    // A layout: row = lane&15. half0: e -> K=e
    //                          half1: e -> K=16+e
    v16h Ag[4];
    #pragma unroll
    for (int g = 0; g < 4; ++g) {
        #pragma unroll
        for (int e = 0; e < 16; ++e) {
            const int k = half * 16 + e;
            float v = 0.0f;
            if (m < STATE) {                      // output row j = m
                const int col = g * STATE + m;    // gate column in [W|U]
                if (k < EMBED)              v = W[k * FOURS + col];
                else if (k - EMBED < STATE) v = U[(k - EMBED) * FOURS + col];
            }
            Ag[g][e] = (_Float16)v;
        }
    }

    // ---- constant C operands: bias broadcast, D row j = VGPR r (lanes 0-15) ----
    v8f Cg[4];
    #pragma unroll
    for (int g = 0; g < 4; ++g)
        #pragma unroll
        for (int r = 0; r < 8; ++r)
            Cg[g][r] = (half == 0 && r < STATE) ? b[g * STATE + r] : 0.0f;

    // ---- dense weights: uniform -> scalar regs ----
    float wd[STATE][NOUT];
    #pragma unroll
    for (int j = 0; j < STATE; ++j)
        #pragma unroll
        for (int o = 0; o < NOUT; ++o)
            wd[j][o] = Wd[j * NOUT + o];
    float bdr[NOUT];
    #pragma unroll
    for (int o = 0; o < NOUT; ++o) bdr[o] = bd[o];

    // ---- recurrent state: all lane-local ----
    float cst[STATE];
    #pragma unroll
    for (int j = 0; j < STATE; ++j) cst[j] = 0.0f;
    unsigned int pk[4] = {0u, 0u, 0u, 0u};  // h packed f16 pairs (K=16..23 dwords)

    const int    tok_base = (b0 + m) * TSTEPS;
    const size_t out_row  = (size_t)(b0 + m) * TSTEPS;

    int tok_next = tokens[tok_base];  // software-pipelined token fetch

    for (int t = 0; t < TSTEPS; ++t) {
        const int tok = tok_next;
        if (t + 1 < TSTEPS) tok_next = tokens[tok_base + t + 1];
        if ((t & 15) == 0 && t + 32 < TSTEPS)
            __builtin_prefetch(&tokens[tok_base + t + 32], 0, 1);

        // ---- B operand = [x^T ; h^T] (32K x 16N), N = batch row = lane&15 ----
        // lanes 0-15: elems 0..15 -> K=0..15 = x ; lanes 16-31: elems 0..6 -> h, rest 0
        const uint4* xr = (const uint4*)&s_emb[tok * EMBED];
        const uint4  x0 = xr[0];
        const uint4  x1 = xr[1];
        B16u bb;
        bb.u[0] = half ? pk[0] : x0.x;
        bb.u[1] = half ? pk[1] : x0.y;
        bb.u[2] = half ? pk[2] : x0.z;
        bb.u[3] = half ? pk[3] : x0.w;
        bb.u[4] = half ? 0u    : x1.x;
        bb.u[5] = half ? 0u    : x1.y;
        bb.u[6] = half ? 0u    : x1.z;
        bb.u[7] = half ? 0u    : x1.w;

        // ---- Z_g^T = A_g x B + bias : lane m holds z[m, j] in VGPR j ----
        v8f zi = __builtin_amdgcn_wmma_f32_16x16x32_f16(false, Ag[0], false, bb.h, (short)0, Cg[0], false, false);
        v8f zf = __builtin_amdgcn_wmma_f32_16x16x32_f16(false, Ag[1], false, bb.h, (short)0, Cg[1], false, false);
        v8f zg = __builtin_amdgcn_wmma_f32_16x16x32_f16(false, Ag[2], false, bb.h, (short)0, Cg[2], false, false);
        v8f zo = __builtin_amdgcn_wmma_f32_16x16x32_f16(false, Ag[3], false, bb.h, (short)0, Cg[3], false, false);

        // ---- LSTM elementwise update: fully per-lane, 7 states ----
        float hv[STATE];
        #pragma unroll
        for (int j = 0; j < STATE; ++j) {
            const float i_ = sigmoid_f(zi[j]);
            const float f_ = sigmoid_f(zf[j]);
            const float g_ = tanh_f   (zg[j]);
            const float o_ = sigmoid_f(zo[j]);
            const float cn = f_ * cst[j] + i_ * g_;
            cst[j] = cn;
            hv[j]  = o_ * tanh_f(cn);
        }

        // ---- h -> f16 pairs, broadcast lane m -> lane m+16 (no LDS memory) ----
        unsigned int q0 = pack2h(hv[0], hv[1]);
        unsigned int q1 = pack2h(hv[2], hv[3]);
        unsigned int q2 = pack2h(hv[4], hv[5]);
        unsigned int q3 = pack2h(hv[6], 0.0f);
        pk[0] = (unsigned int)__shfl((int)q0, m, 32);
        pk[1] = (unsigned int)__shfl((int)q1, m, 32);
        pk[2] = (unsigned int)__shfl((int)q2, m, 32);
        pk[3] = (unsigned int)__shfl((int)q3, m, 32);

        // ---- dense(7->10) + softmax, h already lane-resident (lanes 0-15) ----
        if (half == 0) {
            float logit[NOUT];
            #pragma unroll
            for (int o = 0; o < NOUT; ++o) {
                float acc = bdr[o];
                #pragma unroll
                for (int j = 0; j < STATE; ++j) acc += hv[j] * wd[j][o];
                logit[o] = acc;
            }
            float mx = logit[0];
            #pragma unroll
            for (int o = 1; o < NOUT; ++o) mx = fmaxf(mx, logit[o]);
            float ex[NOUT], s = 0.0f;
            #pragma unroll
            for (int o = 0; o < NOUT; ++o) { ex[o] = __expf(logit[o] - mx); s += ex[o]; }
            const float inv = rcp_f(s);
            float* po = out + (out_row + t) * NOUT;
            float2* po2 = (float2*)po;  // 8-byte aligned (40B stride)
            #pragma unroll
            for (int o = 0; o < NOUT / 2; ++o)
                po2[o] = make_float2(ex[2 * o] * inv, ex[2 * o + 1] * inv);
        }
        // reconverged: next iteration's WMMAs see EXEC all-ones
    }
}

extern "C" void kernel_launch(void* const* d_in, const int* in_sizes, int n_in,
                              void* d_out, int out_size, void* d_ws, size_t ws_size,
                              hipStream_t stream) {
    const int*   tokens = (const int*)  d_in[0];
    const float* emb    = (const float*)d_in[1];
    const float* W      = (const float*)d_in[2];
    const float* U      = (const float*)d_in[3];
    const float* b      = (const float*)d_in[4];
    const float* Wd     = (const float*)d_in[5];
    const float* bd     = (const float*)d_in[6];
    float*       out    = (float*)d_out;

    dim3 grid(BATCH / 16);  // 256 independent recurrence chains
    dim3 block(32);         // one wave32 per tile
    hipLaunchKernelGGL(lstm_wmma_kernel, grid, block, 0, stream,
                       tokens, emb, W, U, b, Wd, bd, out);
}
